// DeformableNet_76218489635492
// MI455X (gfx1250) — compile-verified, matrix operand order checked by reading, not run on hardware
//
#include <hip/hip_runtime.h>
#include <hip/hip_bf16.h>

// ---------------------------------------------------------------------------
// CDNA5 WMMA types / fragment helpers (wave32, 16x16x32 bf16 -> f32)
// ---------------------------------------------------------------------------
typedef __attribute__((ext_vector_type(16))) __bf16 v16bf;
typedef __attribute__((ext_vector_type(8)))  float  v8f;

union FragB { v16bf v; __bf16 h[16]; unsigned u[8]; };
union FragC { v8f   v; float    f[8]; };

// native bf16 convert (hardware cvt on gfx1250, no integer emulation)
__device__ __forceinline__ unsigned pack2(float lo, float hi) {
  union { __bf16 h[2]; unsigned u; } p;
  p.h[0] = (__bf16)lo; p.h[1] = (__bf16)hi;
  return p.u;
}
// 16-bit A-matrix 16x32 layout: lane half picks K-halves, VGPR v holds a K pair.
__device__ __forceinline__ int kidx(int v, int hf) {
  return (v < 4) ? (hf * 8 + 2 * v) : (16 + hf * 8 + 2 * (v - 4));
}
__device__ __forceinline__ float lrelu(float x) { return x > 0.f ? x : 0.2f * x; }
__device__ __forceinline__ int iclamp(int x, int lo, int hi) {
  return x < lo ? lo : (x > hi ? hi : x);
}

// ---------------------------------------------------------------------------
// Stage 0: conv3x3 (Cin=2: fixed||moving) + BN + LeakyReLU.
// 9 taps x 2 ch = 18 <= 32  ->  ONE wmma per 16-channel output tile.
// Each wave: 16 consecutive pixels (one row) x 32 out channels.
// Halo/tail handled by clamped addressing + value select (no exec branches).
// ---------------------------------------------------------------------------
__global__ __launch_bounds__(256) void k_stage0(
    const float* __restrict__ fixedI, const float* __restrict__ movingI,
    const float* __restrict__ w0, const float* __restrict__ g,
    const float* __restrict__ bb, float* __restrict__ out,
    int B, int H, int W)
{
  int wave = blockIdx.x * 8 + (threadIdx.x >> 5);
  int lane = threadIdx.x & 31;
  int tilesX = W >> 4;
  int tx = wave % tilesX;
  int yy = (wave / tilesX) % H;
  int b  = wave / (tilesX * H);
  int x0 = tx << 4;
  int hf = lane >> 4, mr = lane & 15;
  int px = x0 + mr;
  const size_t plane = (size_t)H * W;
  const float* fb = fixedI  + (size_t)b * plane;
  const float* mb = movingI + (size_t)b * plane;

  FragB a;
#pragma unroll
  for (int v = 0; v < 8; ++v) {
    int k0 = kidx(v, hf);          // even; k0 -> fixed, k0+1 -> moving, same tap
    bool valid = (k0 < 18);
    int tap = valid ? (k0 >> 1) : 0;
    int y = yy + tap / 3 - 1, x = px + tap % 3 - 1;
    bool inb = valid && y >= 0 && y < H && x >= 0 && x < W;
    size_t off = (size_t)iclamp(y, 0, H - 1) * W + iclamp(x, 0, W - 1);
    float lo = fb[off];
    float hi = mb[off];
    a.h[2 * v]     = (__bf16)(inb ? lo : 0.f);
    a.h[2 * v + 1] = (__bf16)(inb ? hi : 0.f);
  }

  FragC acc0, acc1;
#pragma unroll
  for (int v = 0; v < 8; ++v) { acc0.f[v] = 0.f; acc1.f[v] = 0.f; }

#pragma unroll
  for (int nt = 0; nt < 2; ++nt) {
    int n = nt * 16 + mr;
    FragB wf;
#pragma unroll
    for (int v = 0; v < 8; ++v) {
      int k0 = kidx(v, hf);
      int k0c = k0 < 18 ? k0 : 0;
      float lo = w0[k0c * 32 + n];
      float hi = (k0 + 1 < 18) ? w0[(k0 + 1) * 32 + n] : 0.f;
      wf.h[2 * v]     = (__bf16)(k0 < 18 ? lo : 0.f);
      wf.h[2 * v + 1] = (__bf16)hi;
    }
    if (nt == 0)
      acc0.v = __builtin_amdgcn_wmma_f32_16x16x32_bf16(false, a.v, false, wf.v,
                                                       (short)0, acc0.v, false, false);
    else
      acc1.v = __builtin_amdgcn_wmma_f32_16x16x32_bf16(false, a.v, false, wf.v,
                                                       (short)0, acc1.v, false, false);
  }

  int n0 = mr, n1 = 16 + mr;
  float s0 = g[n0] * rsqrtf(1.0f + 1e-3f), s1 = g[n1] * rsqrtf(1.0f + 1e-3f);
  float b0 = bb[n0], b1 = bb[n1];
  float* ob = out + (((size_t)b * H + yy) * W) * 32;
#pragma unroll
  for (int v = 0; v < 8; ++v) {
    int mrow = v + 8 * hf;                       // D layout: m = v + 8*half
    size_t o = (size_t)(x0 + mrow) * 32;
    ob[o + n0] = lrelu(acc0.f[v] * s0 + b0);
    ob[o + n1] = lrelu(acc1.f[v] * s1 + b1);
  }
}

// ---------------------------------------------------------------------------
// Generic conv3x3 Cin=32 -> Cout=32 + BN + LeakyReLU.
// Weights pre-packed once per block into LDS as ready bf16x2 B-fragments.
// Each wave: 16 pixels x 32 out channels = 9 taps x 2 WMMAs.
// Halo: clamped addresses + value select -> unconditional b64 load clauses.
// ---------------------------------------------------------------------------
__global__ __launch_bounds__(256) void k_conv3x3_c32(
    const float* __restrict__ in, const float* __restrict__ w,
    const float* __restrict__ g, const float* __restrict__ bb,
    float* __restrict__ out, int B, int H, int W)
{
  __shared__ unsigned wfrag[9][2][8][32];        // [tap][ntile][vgpr][lane], 18KB
  const int tid = threadIdx.x;
  for (int e = tid; e < 9 * 2 * 8 * 32; e += 256) {
    int lane = e & 31;
    int v    = (e >> 5) & 7;
    int nt   = (e >> 8) & 1;
    int tap  = e >> 9;
    int hf = lane >> 4;
    int n  = nt * 16 + (lane & 15);
    int k0 = kidx(v, hf);
    wfrag[tap][nt][v][lane] =
        pack2(w[(tap * 32 + k0) * 32 + n], w[(tap * 32 + k0 + 1) * 32 + n]);
  }
  __syncthreads();

  int wave = blockIdx.x * 8 + (tid >> 5);
  int lane = tid & 31;
  int tilesX = W >> 4;
  int tx = wave % tilesX;
  int yy = (wave / tilesX) % H;
  int b  = wave / (tilesX * H);
  int x0 = tx << 4;
  int hf = lane >> 4, mr = lane & 15;
  int px = x0 + mr;
  const float* inB = in + (size_t)b * H * W * 32;

  FragC acc0, acc1;
#pragma unroll
  for (int v = 0; v < 8; ++v) { acc0.f[v] = 0.f; acc1.f[v] = 0.f; }

#pragma unroll
  for (int tap = 0; tap < 9; ++tap) {
    int iy = yy + tap / 3 - 1;
    int ix = px + tap % 3 - 1;
    bool inb = (iy >= 0 && iy < H && ix >= 0 && ix < W);
    size_t off = ((size_t)iclamp(iy, 0, H - 1) * W + iclamp(ix, 0, W - 1)) * 32;
    const float2* s2 = (const float2*)(inB + off);   // 8B aligned (k0 even)
    FragB a;
#pragma unroll
    for (int v = 0; v < 8; ++v) {
      int k0 = kidx(v, hf);
      float2 p = s2[k0 >> 1];
      a.h[2 * v]     = (__bf16)(inb ? p.x : 0.f);
      a.h[2 * v + 1] = (__bf16)(inb ? p.y : 0.f);
    }
    FragB b0, b1;
#pragma unroll
    for (int v = 0; v < 8; ++v) { b0.u[v] = wfrag[tap][0][v][lane];
                                  b1.u[v] = wfrag[tap][1][v][lane]; }
    acc0.v = __builtin_amdgcn_wmma_f32_16x16x32_bf16(false, a.v, false, b0.v,
                                                     (short)0, acc0.v, false, false);
    acc1.v = __builtin_amdgcn_wmma_f32_16x16x32_bf16(false, a.v, false, b1.v,
                                                     (short)0, acc1.v, false, false);
  }

  int n0 = mr, n1 = 16 + mr;
  float s0 = g[n0] * rsqrtf(1.0f + 1e-3f), s1 = g[n1] * rsqrtf(1.0f + 1e-3f);
  float bi0 = bb[n0], bi1 = bb[n1];
  float* ob = out + (((size_t)b * H + yy) * W) * 32;
#pragma unroll
  for (int v = 0; v < 8; ++v) {
    int mrow = v + 8 * hf;
    size_t o = (size_t)(x0 + mrow) * 32;
    ob[o + n0] = lrelu(acc0.f[v] * s0 + bi0);
    ob[o + n1] = lrelu(acc1.f[v] * s1 + bi1);
  }
}

// ---------------------------------------------------------------------------
// 2x2 average pool, stride 2, C=32 (NHWC).
// ---------------------------------------------------------------------------
__global__ __launch_bounds__(256) void k_pool(
    const float* __restrict__ in, float* __restrict__ out, int B, int H, int W)
{
  int idx = blockIdx.x * 256 + threadIdx.x;
  int Ho = H >> 1, Wo = W >> 1;
  int total = B * Ho * Wo * 32;
  if (idx >= total) return;
  int c = idx & 31; int t = idx >> 5;
  int x = t % Wo; t /= Wo;
  int y = t % Ho; int b = t / Ho;
  const float* p = in + (((size_t)b * H + 2 * y) * W + 2 * x) * 32 + c;
  size_t rs = (size_t)W * 32;
  out[idx] = 0.25f * (p[0] + p[32] + p[rs] + p[rs + 32]);
}

// ---------------------------------------------------------------------------
// 1x1 conv (Cin=32, Cout = 32 or 2). mode 0: BN + LeakyReLU; mode 1: +bias only.
// ---------------------------------------------------------------------------
__global__ __launch_bounds__(256) void k_pw(
    const float* __restrict__ in, const float* __restrict__ w,
    const float* __restrict__ g, const float* __restrict__ bb,
    float* __restrict__ out, int cout, int mode)
{
  int wave = blockIdx.x * 8 + (threadIdx.x >> 5);
  int lane = threadIdx.x & 31;
  int hf = lane >> 4, mr = lane & 15;
  size_t pixbase = (size_t)wave * 16;
  const float2* src = (const float2*)(in + (pixbase + mr) * 32);

  FragB a;
#pragma unroll
  for (int v = 0; v < 8; ++v) {
    int k0 = kidx(v, hf);
    float2 p = src[k0 >> 1];
    a.h[2 * v]     = (__bf16)p.x;
    a.h[2 * v + 1] = (__bf16)p.y;
  }
  int ntiles = (cout + 15) >> 4;
  for (int nt = 0; nt < ntiles; ++nt) {
    int n = nt * 16 + mr;
    int nc = n < cout ? n : 0;
    FragB wf;
#pragma unroll
    for (int v = 0; v < 8; ++v) {
      int k0 = kidx(v, hf);
      float lo = w[k0 * cout + nc];
      float hi = w[(k0 + 1) * cout + nc];
      wf.h[2 * v]     = (__bf16)(n < cout ? lo : 0.f);
      wf.h[2 * v + 1] = (__bf16)(n < cout ? hi : 0.f);
    }
    FragC acc;
#pragma unroll
    for (int v = 0; v < 8; ++v) acc.f[v] = 0.f;
    acc.v = __builtin_amdgcn_wmma_f32_16x16x32_bf16(false, a.v, false, wf.v,
                                                    (short)0, acc.v, false, false);
    if (n < cout) {
      float sc = (mode == 0) ? g[n] * rsqrtf(1.0f + 1e-3f) : 1.f;
      float bi = bb[n];
#pragma unroll
      for (int v = 0; v < 8; ++v) {
        int mrow = v + 8 * hf;
        float r = acc.f[v] * sc + bi;
        if (mode == 0) r = lrelu(r);
        out[(pixbase + mrow) * cout + n] = r;
      }
    }
  }
}

// ---------------------------------------------------------------------------
// Fused B-spline transposed conv (stride 8, K=32), bending-energy penalty,
// and bilinear warp. Per output pixel: 4x4 control-point taps; separable
// outer-product kernels evaluated analytically.
// ---------------------------------------------------------------------------
__device__ __forceinline__ void bspline_basis(int t, float& kv, float& dkv, float& ddkv)
{
  // flat kernel: k[c*8+u] = basis_c evaluated at (7-u)/8 (see reference flat())
  const float C0[4] = {-1.f/6.f,  3.f/6.f, -3.f/6.f, 1.f/6.f};
  const float C1[4] = { 3.f/6.f, -6.f/6.f,  3.f/6.f, 0.f};
  const float C2[4] = {-3.f/6.f,  0.f,      3.f/6.f, 0.f};
  const float C3[4] = { 1.f/6.f,  4.f/6.f,  1.f/6.f, 0.f};
  int c = t >> 3, u = t & 7;
  float tv = (float)(7 - u) * 0.125f;
  kv   = ((C0[c]*tv + C1[c])*tv + C2[c])*tv + C3[c];
  dkv  = (3.f*C0[c]*tv + 2.f*C1[c])*tv + C2[c];
  ddkv = 6.f*C0[c]*tv + 2.f*C1[c];
}

__global__ __launch_bounds__(256) void k_bspline_warp(
    const float* __restrict__ disp, const float* __restrict__ moving,
    float* __restrict__ out, float* __restrict__ part, int B, int H, int W)
{
  int idx = blockIdx.x * 256 + threadIdx.x;
  int total = B * H * W;
  float myPen = 0.f;
  if (idx < total) {
    int x = idx % W; int t1 = idx / W; int y = t1 % H; int b = t1 / H;
    int HD = H >> 3, WD = W >> 3;
    // valid taps: k2d index t = o - 8i + 12 in [0,32)  ->  i = ihi - j, t = t0 + 8j
    int ihx = (x + 12) >> 3, t0x = (x + 12) & 7;
    int ihy = (y + 12) >> 3, t0y = (y + 12) & 7;
    float kx[4], dkx[4], ddkx[4], ky[4], dky[4], ddky[4];
#pragma unroll
    for (int j = 0; j < 4; ++j) {
      bspline_basis(t0x + 8 * j, kx[j], dkx[j], ddkx[j]);
      bspline_basis(t0y + 8 * j, ky[j], dky[j], ddky[j]);
    }
    float up0=0.f, up1=0.f, xx0=0.f, xx1=0.f, yy0=0.f, yy1=0.f, xy0=0.f, xy1=0.f;
    const float* db = disp + (size_t)b * HD * WD * 2;
#pragma unroll
    for (int jy = 0; jy < 4; ++jy) {
      int iy = ihy - jy;
      if (iy < 0 || iy >= HD) continue;
#pragma unroll
      for (int jx = 0; jx < 4; ++jx) {
        int ix = ihx - jx;
        if (ix < 0 || ix >= WD) continue;
        const float* dp = db + ((size_t)iy * WD + ix) * 2;
        float d0 = dp[0], d1 = dp[1];
        float wk  = ky[jy]  * kx[jx];
        float wxx = ky[jy]  * ddkx[jx];   // Kxx = outer(k, ddk)
        float wyy = ddky[jy]* kx[jx];     // Kyy = outer(ddk, k)
        float wxy = dky[jy] * dkx[jx];    // Kxy = outer(dk, dk)
        up0 += d0 * wk;  up1 += d1 * wk;
        xx0 += d0 * wxx; xx1 += d1 * wxx;
        yy0 += d0 * wyy; yy1 += d1 * wyy;
        xy0 += d0 * wxy; xy1 += d1 * wxy;
      }
    }
    myPen = (xx0*xx0 + xx1*xx1 + yy0*yy0 + yy1*yy1 + 2.f*(xy0*xy0 + xy1*xy1))
            / (float)B;

    // bilinear warp (matches reference clamp/eps semantics)
    float x_t = up0 + (float)x;
    float y_t = up1 + (float)y;
    float fx = floorf(x_t), fy = floorf(y_t);
    float Wm1 = (float)(W - 1), Hm1 = (float)(H - 1);
    float x0c = fminf(fmaxf(fx,       0.f), Wm1);
    float y0c = fminf(fmaxf(fy,       0.f), Hm1);
    float x1c = fminf(fmaxf(fx + 1.f, 0.f), Wm1);
    float y1c = fminf(fmaxf(fy + 1.f, 0.f), Hm1);
    const float* img = moving + (size_t)b * H * W;
    float Q1 = img[(int)(y0c * (float)W + x0c)];
    float Q2 = img[(int)(y1c * (float)W + x0c)];
    float Q3 = img[(int)(y0c * (float)W + x1c)];
    float Q4 = img[(int)(y1c * (float)W + x1c)];
    const float eps = 1e-5f;
    float wa = (x1c - x_t) / (x1c - x0c + eps);
    float wb = (x_t - x0c) / (x1c - x0c + eps);
    float R1 = wa * Q1 + wb * Q3;
    float R2 = wa * Q2 + wb * Q4;
    out[idx] = (y1c - y_t) / (y1c - y0c + eps) * R1
             + (y_t - y0c) / (y1c - y0c + eps) * R2;
  }
  // deterministic block reduction of penalty partials
  __shared__ float red[256];
  red[threadIdx.x] = myPen;
  __syncthreads();
  for (int s = 128; s > 0; s >>= 1) {
    if (threadIdx.x < s) red[threadIdx.x] += red[threadIdx.x + s];
    __syncthreads();
  }
  if (threadIdx.x == 0) part[blockIdx.x] = red[0];
}

__global__ __launch_bounds__(256) void k_reduce(
    const float* __restrict__ part, int n, float* __restrict__ pen)
{
  __shared__ float red[256];
  float s = 0.f;
  for (int i = threadIdx.x; i < n; i += 256) s += part[i];
  red[threadIdx.x] = s;
  __syncthreads();
  for (int st = 128; st > 0; st >>= 1) {
    if (threadIdx.x < st) red[threadIdx.x] += red[threadIdx.x + st];
    __syncthreads();
  }
  if (threadIdx.x == 0) *pen = red[0];
}

// ---------------------------------------------------------------------------
// Host-side orchestration
// ---------------------------------------------------------------------------
extern "C" void kernel_launch(void* const* d_in, const int* in_sizes, int n_in,
                              void* d_out, int out_size, void* d_ws, size_t ws_size,
                              hipStream_t stream)
{
  (void)in_sizes; (void)n_in; (void)out_size; (void)ws_size;
  const int B = 8, H = 512, W = 512, C = 32;

  const float* fixedI  = (const float*)d_in[0];
  const float* movingI = (const float*)d_in[1];
  const float* sw[3] = {(const float*)d_in[2], (const float*)d_in[3], (const float*)d_in[4]};
  const float* sg[3] = {(const float*)d_in[5], (const float*)d_in[6], (const float*)d_in[7]};
  const float* sb[3] = {(const float*)d_in[8], (const float*)d_in[9], (const float*)d_in[10]};
  const float* fw[2] = {(const float*)d_in[11], (const float*)d_in[12]};
  const float* fg[2] = {(const float*)d_in[13], (const float*)d_in[14]};
  const float* fb[2] = {(const float*)d_in[15], (const float*)d_in[16]};
  const float* pww[2] = {(const float*)d_in[17], (const float*)d_in[18]};
  const float* pwg[2] = {(const float*)d_in[19], (const float*)d_in[20]};
  const float* pwb[2] = {(const float*)d_in[21], (const float*)d_in[22]};
  const float* outw = (const float*)d_in[23];
  const float* outb = (const float*)d_in[24];

  float* buf0 = (float*)d_ws;                         // 8*512*512*32 floats max
  float* buf1 = buf0 + (size_t)B * H * W * C;         // 8*256*256*32 floats max
  float* fout = (float*)d_out;                        // [B,H,W,1] then penalty
  float* pen  = fout + (size_t)B * H * W;

  // stage 0: conv(2->32)+BN+lrelu @512  -> buf0
  k_stage0<<<(B * H * (W / 16)) / 8, 256, 0, stream>>>(
      fixedI, movingI, sw[0], sg[0], sb[0], buf0, B, H, W);
  // pool -> buf1 @256
  k_pool<<<(B * (H/2) * (W/2) * 32 + 255) / 256, 256, 0, stream>>>(buf0, buf1, B, H, W);
  // stage 1 @256 -> buf0
  k_conv3x3_c32<<<(B * (H/2) * ((W/2)/16)) / 8, 256, 0, stream>>>(
      buf1, sw[1], sg[1], sb[1], buf0, B, H/2, W/2);
  // pool -> buf1 @128
  k_pool<<<(B * (H/4) * (W/4) * 32 + 255) / 256, 256, 0, stream>>>(buf0, buf1, B, H/2, W/2);
  // stage 2 @128 -> buf0
  k_conv3x3_c32<<<(B * (H/4) * ((W/4)/16)) / 8, 256, 0, stream>>>(
      buf1, sw[2], sg[2], sb[2], buf0, B, H/4, W/4);
  // pool -> buf1 @64
  k_pool<<<(B * (H/8) * (W/8) * 32 + 255) / 256, 256, 0, stream>>>(buf0, buf1, B, H/4, W/4);
  // final conv 1 @64 -> buf0
  k_conv3x3_c32<<<(B * (H/8) * ((W/8)/16)) / 8, 256, 0, stream>>>(
      buf1, fw[0], fg[0], fb[0], buf0, B, H/8, W/8);
  // final conv 2 @64 -> buf1
  k_conv3x3_c32<<<(B * (H/8) * ((W/8)/16)) / 8, 256, 0, stream>>>(
      buf0, fw[1], fg[1], fb[1], buf1, B, H/8, W/8);
  // pointwise convs @64
  const int npix = B * (H/8) * (W/8);                 // 32768
  k_pw<<<(npix / 16) / 8, 256, 0, stream>>>(buf1, pww[0], pwg[0], pwb[0], buf0, 32, 0);
  k_pw<<<(npix / 16) / 8, 256, 0, stream>>>(buf0, pww[1], pwg[1], pwb[1], buf1, 32, 0);
  // output head (32->2, bias only) -> disp in buf0
  k_pw<<<(npix / 16) / 8, 256, 0, stream>>>(buf1, outw, outb, outb, buf0, 2, 1);

  // fused B-spline upsample + penalty partials (buf1 reused for partials) + warp
  const int nblocks = (B * H * W) / 256;              // 8192
  k_bspline_warp<<<nblocks, 256, 0, stream>>>(buf0, movingI, fout, buf1, B, H, W);
  k_reduce<<<1, 256, 0, stream>>>(buf1, nblocks, pen);
}
